// MultiheadedAttentionV2_17248588661234
// MI455X (gfx1250) — compile-verified, matrix operand order checked by reading, not run on hardware
//
#include <hip/hip_runtime.h>
#include <hip/hip_bf16.h>

// MI455X / gfx1250 fused multi-head attention (compile-only target).
// f16 WMMA (v_wmma_f32_16x16x32_f16) for all four matmuls, flash-attention
// fusion (S x S scores never hit HBM), and CDNA5 async LDS staging:
// K/V are pre-converted to f16 (V pre-transposed), then the attention hot
// loop stages tiles with GLOBAL_LOAD_ASYNC_TO_LDS_B128 (ASYNCcnt) into
// double-buffered LDS, overlapping copy with WMMA compute.

#define EMB   512
#define HEADS 8
#define KD    64
#define SEQ   2048
#define BATCH 4

typedef __attribute__((ext_vector_type(16))) _Float16 v16h;
typedef __attribute__((ext_vector_type(8)))  _Float16 v8h;
typedef __attribute__((ext_vector_type(8)))  float    v8f;

static __device__ __forceinline__ v8f wmma_f16(v16h a, v16h b, v8f c) {
  // (neg_a, A, neg_b, B, c_mod, C, reuse_a, reuse_b)
  return __builtin_amdgcn_wmma_f32_16x16x32_f16(false, a, false, b, (short)0, c,
                                                false, false);
}

// Build a 16-half A/B fragment from two contiguous 8-half (16B) chunks.
static __device__ __forceinline__ v16h frag_from_h(const _Float16* p0,
                                                   const _Float16* p1) {
  v8h lo = *(const v8h*)p0;
  v8h hi = *(const v8h*)p1;
  return __builtin_shufflevector(lo, hi, 0, 1, 2, 3, 4, 5, 6, 7, 8, 9, 10, 11,
                                 12, 13, 14, 15);
}

// Build a fragment from two contiguous 8-float chunks (convert f32 -> f16).
static __device__ __forceinline__ v16h frag_from_f32(const float* p0,
                                                     const float* p1) {
  float4 a0 = *(const float4*)(p0);
  float4 a1 = *(const float4*)(p0 + 4);
  float4 b0 = *(const float4*)(p1);
  float4 b1 = *(const float4*)(p1 + 4);
  v16h r;
  r[0]  = (_Float16)a0.x; r[1]  = (_Float16)a0.y;
  r[2]  = (_Float16)a0.z; r[3]  = (_Float16)a0.w;
  r[4]  = (_Float16)a1.x; r[5]  = (_Float16)a1.y;
  r[6]  = (_Float16)a1.z; r[7]  = (_Float16)a1.w;
  r[8]  = (_Float16)b0.x; r[9]  = (_Float16)b0.y;
  r[10] = (_Float16)b0.z; r[11] = (_Float16)b0.w;
  r[12] = (_Float16)b1.x; r[13] = (_Float16)b1.y;
  r[14] = (_Float16)b1.z; r[15] = (_Float16)b1.w;
  return r;
}

static __device__ __forceinline__ v8h pack8(float4 a, float4 b) {
  v8h h;
  h[0] = (_Float16)a.x; h[1] = (_Float16)a.y;
  h[2] = (_Float16)a.z; h[3] = (_Float16)a.w;
  h[4] = (_Float16)b.x; h[5] = (_Float16)b.y;
  h[6] = (_Float16)b.z; h[7] = (_Float16)b.w;
  return h;
}

// Issue one per-lane 16B async global->LDS copy (ASYNCcnt tracked).
static __device__ __forceinline__ void async_copy16(const _Float16* g,
                                                    _Float16* lds) {
  unsigned ldsoff = (unsigned)(uintptr_t)lds;  // LDS aperture: low 32 bits
  asm volatile("global_load_async_to_lds_b128 %0, %1, off"
               :
               : "v"(ldsoff), "v"((unsigned long long)(uintptr_t)g)
               : "memory");
}
static __device__ __forceinline__ void wait_async0() {
  asm volatile("s_wait_asynccnt 0x0" ::: "memory");
}

// ---------------------------------------------------------------------------
// Kernel 0: K/V precision prep. kh = f16(keys) [B,H,S,64];
// vh = f16(values) transposed to [B,H,64,S] so attention V tiles are
// contiguous per dim-row (pure 16B async copies later).
// ---------------------------------------------------------------------------
__global__ __launch_bounds__(256) void kvprep_kernel(
    const float* __restrict__ keys, const float* __restrict__ values,
    _Float16* __restrict__ kh, _Float16* __restrict__ vh) {
  alignas(16) __shared__ _Float16 tile[64][72];  // 64 dims x 64 seq (+pad)

  const int nst = SEQ / 64;
  const int st  = blockIdx.x % nst;
  const int bh  = blockIdx.x / nst;
  const int s0  = st * 64;
  const int t   = threadIdx.x;
  const int r   = t >> 2;            // local seq row 0..63
  const int d0  = (t & 3) * 16;      // dim chunk

  const float* kp = keys   + ((size_t)bh * SEQ + s0 + r) * KD + d0;
  const float* vp = values + ((size_t)bh * SEQ + s0 + r) * KD + d0;
  _Float16*    ko = kh     + ((size_t)bh * SEQ + s0 + r) * KD + d0;

#pragma unroll
  for (int i = 0; i < 2; ++i) {
    float4 a = ((const float4*)kp)[2 * i];
    float4 b = ((const float4*)kp)[2 * i + 1];
    ((v8h*)ko)[i] = pack8(a, b);
  }
#pragma unroll
  for (int i = 0; i < 4; ++i) {
    float4 a = ((const float4*)vp)[i];
    tile[d0 + 4 * i + 0][r] = (_Float16)a.x;
    tile[d0 + 4 * i + 1][r] = (_Float16)a.y;
    tile[d0 + 4 * i + 2][r] = (_Float16)a.z;
    tile[d0 + 4 * i + 3][r] = (_Float16)a.w;
  }
  __syncthreads();

  const int d  = t >> 2;             // dim 0..63
  const int sc = (t & 3) * 16;       // seq chunk
  _Float16* vo = vh + ((size_t)bh * KD + d) * SEQ + s0 + sc;
#pragma unroll
  for (int i = 0; i < 2; ++i) {
    ((v8h*)vo)[i] = *(const v8h*)&tile[d][sc + 8 * i];
  }
}

// ---------------------------------------------------------------------------
// Kernel 1: Q projection. Q = (x @ Wq) * (1/sqrt(64)), output f16 [B,H,S,64].
// ---------------------------------------------------------------------------
__global__ __launch_bounds__(128) void qproj_kernel(
    const float* __restrict__ x, const float* __restrict__ wq,
    _Float16* __restrict__ qout) {
  alignas(16) __shared__ _Float16 ldsB[16][32];  // [n][k] (transposed tile)

  const int nt    = blockIdx.x;          // 0..31  (N tile of 16 cols)
  const int mtg   = blockIdx.y;          // 0..127 (group of 4 M tiles)
  const int wave  = threadIdx.x >> 5;
  const int lane  = threadIdx.x & 31;
  const int half  = lane >> 4;
  const int l16   = lane & 15;
  const int mrow  = (mtg * 4 + wave) * 16;
  const int ncol0 = nt * 16;

  const int sk = threadIdx.x & 31;         // staging: k within tile
  const int sn = (threadIdx.x >> 5) * 4;   // staging: n group of 4

  const float* xrow = x + (size_t)(mrow + l16) * EMB;
  v8f acc = {};

#pragma unroll 2
  for (int k0 = 0; k0 < EMB; k0 += 32) {
    float4 wv = *(const float4*)&wq[(size_t)(k0 + sk) * EMB + ncol0 + sn];
    __syncthreads();
    ldsB[sn + 0][sk] = (_Float16)wv.x;
    ldsB[sn + 1][sk] = (_Float16)wv.y;
    ldsB[sn + 2][sk] = (_Float16)wv.z;
    ldsB[sn + 3][sk] = (_Float16)wv.w;
    __syncthreads();

    v16h a = frag_from_f32(xrow + k0 + half * 8, xrow + k0 + 16 + half * 8);
    v16h b = frag_from_h(&ldsB[l16][half * 8], &ldsB[l16][16 + half * 8]);
    acc = wmma_f16(a, b, acc);
  }

  const int hh = ncol0 >> 6;   // head
  const int d0 = ncol0 & 63;   // dim offset within head
#pragma unroll
  for (int r = 0; r < 8; ++r) {
    int grow = mrow + half * 8 + r;      // flat row in [B*S]
    int bb = grow >> 11;
    int ss = grow & (SEQ - 1);
    qout[(((size_t)bb * HEADS + hh) * SEQ + ss) * KD + d0 + l16] =
        (_Float16)(acc[r] * 0.125f);
  }
}

// ---------------------------------------------------------------------------
// Kernel 2: fused flash attention with async double-buffered K/V staging.
// One block per (b, h, 64 query rows); 4 waves each own a 16-row query tile.
// ---------------------------------------------------------------------------
__global__ __launch_bounds__(128) void attn_kernel(
    const _Float16* __restrict__ kh, const _Float16* __restrict__ vh,
    const _Float16* __restrict__ q, _Float16* __restrict__ ctx) {
  alignas(16) __shared__ _Float16 Kt[2][32][64];   // [buf][key][dim]
  alignas(16) __shared__ _Float16 Vt[2][64][32];   // [buf][dim][key]
  alignas(16) __shared__ _Float16 Pb[4][16][32];   // per-wave P relayout

  const int nqb  = SEQ / 64;
  const int qb   = blockIdx.x % nqb;
  const int bh   = blockIdx.x / nqb;         // b*HEADS + h
  const int wave = threadIdx.x >> 5;
  const int lane = threadIdx.x & 31;
  const int half = lane >> 4;
  const int l16  = lane & 15;
  const int q0   = qb * 64 + wave * 16;
  const int t    = threadIdx.x;

  const _Float16* kbase = kh + (size_t)bh * SEQ * KD;   // [S][64]
  const _Float16* vbase = vh + (size_t)bh * KD * SEQ;   // [64][S]
  const _Float16* qbase = q  + (size_t)bh * SEQ * KD;

  // Q A-fragments (pre-scaled by 1/8 in qproj): dims 0..31 and 32..63.
  const _Float16* qrow = qbase + (size_t)(q0 + l16) * KD;
  v16h qa0 = frag_from_h(qrow + half * 8,      qrow + 16 + half * 8);
  v16h qa1 = frag_from_h(qrow + 32 + half * 8, qrow + 48 + half * 8);

  v8f acc0 = {}, acc1 = {}, acc2 = {}, acc3 = {};
  float m[8], l[8];
#pragma unroll
  for (int r = 0; r < 8; ++r) { m[r] = -1e30f; l[r] = 0.0f; }

  // Async-stage one 32-key tile (K: 4KB, V: 4KB) into buffer p.
  // 128 threads x (2+2) x 16B per-lane async copies.
  auto stage = [&](int j0, int p) {
#pragma unroll
    for (int i = 0; i < 2; ++i) {
      int c = t + i * 128;                 // 0..255
      int key = c >> 3, o = (c & 7) * 8;   // 16B chunk within 128B key row
      async_copy16(kbase + (size_t)(j0 + key) * KD + o, &Kt[p][key][o]);
    }
#pragma unroll
    for (int i = 0; i < 2; ++i) {
      int c = t + i * 128;
      int d = c >> 2, o = (c & 3) * 8;     // 16B chunk within 64B dim row
      async_copy16(vbase + (size_t)d * SEQ + j0 + o, &Vt[p][d][o]);
    }
  };

  stage(0, 0);
  wait_async0();
  __syncthreads();

  int p = 0;
  for (int j0 = 0; j0 < SEQ; j0 += 32) {
    if (j0 + 32 < SEQ) stage(j0 + 32, p ^ 1);  // overlap copy with compute

    // ---- scores: two 16-key subtiles, K-dim 64 = 2 WMMAs each ----
    v8f s0 = {}, s1 = {};
    {
      v16h b00 = frag_from_h(&Kt[p][l16][half * 8],
                             &Kt[p][l16][16 + half * 8]);
      v16h b01 = frag_from_h(&Kt[p][l16][32 + half * 8],
                             &Kt[p][l16][48 + half * 8]);
      s0 = wmma_f16(qa0, b00, s0);
      s0 = wmma_f16(qa1, b01, s0);
      v16h b10 = frag_from_h(&Kt[p][16 + l16][half * 8],
                             &Kt[p][16 + l16][16 + half * 8]);
      v16h b11 = frag_from_h(&Kt[p][16 + l16][32 + half * 8],
                             &Kt[p][16 + l16][48 + half * 8]);
      s1 = wmma_f16(qa0, b10, s1);
      s1 = wmma_f16(qa1, b11, s1);
    }

    // ---- online softmax over the 32 new columns ----
    float rescale[8];
#pragma unroll
    for (int r = 0; r < 8; ++r) {
      float mx = fmaxf(s0[r], s1[r]);
#pragma unroll
      for (int off = 1; off < 16; off <<= 1)
        mx = fmaxf(mx, __shfl_xor(mx, off, 32));   // within 16-lane C group
      float mn = fmaxf(m[r], mx);
      float sc = __expf(m[r] - mn);
      float p0 = __expf(s0[r] - mn);
      float p1 = __expf(s1[r] - mn);
      float sum = p0 + p1;
#pragma unroll
      for (int off = 1; off < 16; off <<= 1) sum += __shfl_xor(sum, off, 32);
      l[r] = l[r] * sc + sum;
      m[r] = mn;
      rescale[r] = sc;
      int row = r + half * 8;                       // C-layout row
      Pb[wave][row][l16]      = (_Float16)p0;       // relayout C -> A via LDS
      Pb[wave][row][16 + l16] = (_Float16)p1;
    }
#pragma unroll
    for (int r = 0; r < 8; ++r) {
      acc0[r] *= rescale[r]; acc1[r] *= rescale[r];
      acc2[r] *= rescale[r]; acc3[r] *= rescale[r];
    }

    // ---- context: P(16x32) @ V(32x64), 4 WMMAs ----
    v16h pa = frag_from_h(&Pb[wave][l16][half * 8],
                          &Pb[wave][l16][16 + half * 8]);
    v16h vb0 = frag_from_h(&Vt[p][l16][half * 8],
                           &Vt[p][l16][16 + half * 8]);
    v16h vb1 = frag_from_h(&Vt[p][16 + l16][half * 8],
                           &Vt[p][16 + l16][16 + half * 8]);
    v16h vb2 = frag_from_h(&Vt[p][32 + l16][half * 8],
                           &Vt[p][32 + l16][16 + half * 8]);
    v16h vb3 = frag_from_h(&Vt[p][48 + l16][half * 8],
                           &Vt[p][48 + l16][16 + half * 8]);
    acc0 = wmma_f16(pa, vb0, acc0);
    acc1 = wmma_f16(pa, vb1, acc1);
    acc2 = wmma_f16(pa, vb2, acc2);
    acc3 = wmma_f16(pa, vb3, acc3);

    wait_async0();     // next tile landed
    __syncthreads();   // all waves done with buffer p / next tile visible
    p ^= 1;
  }

  // Normalize and write ctx as f16 [B*S][HEADS*KD].
  const int bi = bh / HEADS, hi = bh % HEADS;
#pragma unroll
  for (int r = 0; r < 8; ++r) {
    float inv = 1.0f / l[r];
    size_t row = (size_t)bi * SEQ + q0 + r + half * 8;
    _Float16* cp = ctx + row * EMB + hi * KD + l16;
    cp[0]  = (_Float16)(acc0[r] * inv);
    cp[16] = (_Float16)(acc1[r] * inv);
    cp[32] = (_Float16)(acc2[r] * inv);
    cp[48] = (_Float16)(acc3[r] * inv);
  }
}

// ---------------------------------------------------------------------------
// Kernel 3: output projection. out = ctx(f16) @ Wo(f32), f32 output.
// ---------------------------------------------------------------------------
__global__ __launch_bounds__(128) void oproj_kernel(
    const _Float16* __restrict__ ctx, const float* __restrict__ wo,
    float* __restrict__ out) {
  alignas(16) __shared__ _Float16 ldsB[16][32];  // [n][k]

  const int nt    = blockIdx.x;
  const int mtg   = blockIdx.y;
  const int wave  = threadIdx.x >> 5;
  const int lane  = threadIdx.x & 31;
  const int half  = lane >> 4;
  const int l16   = lane & 15;
  const int mrow  = (mtg * 4 + wave) * 16;
  const int ncol0 = nt * 16;

  const int sk = threadIdx.x & 31;
  const int sn = (threadIdx.x >> 5) * 4;

  const _Float16* crow = ctx + (size_t)(mrow + l16) * EMB;
  v8f acc = {};

#pragma unroll 2
  for (int k0 = 0; k0 < EMB; k0 += 32) {
    float4 wv = *(const float4*)&wo[(size_t)(k0 + sk) * EMB + ncol0 + sn];
    __syncthreads();
    ldsB[sn + 0][sk] = (_Float16)wv.x;
    ldsB[sn + 1][sk] = (_Float16)wv.y;
    ldsB[sn + 2][sk] = (_Float16)wv.z;
    ldsB[sn + 3][sk] = (_Float16)wv.w;
    __syncthreads();

    v16h a = frag_from_h(crow + k0 + half * 8, crow + k0 + 16 + half * 8);
    v16h b = frag_from_h(&ldsB[l16][half * 8], &ldsB[l16][16 + half * 8]);
    acc = wmma_f16(a, b, acc);
  }

#pragma unroll
  for (int r = 0; r < 8; ++r) {
    size_t grow = (size_t)(mrow + half * 8 + r);
    out[grow * EMB + ncol0 + l16] = acc[r];
  }
}

// ---------------------------------------------------------------------------
extern "C" void kernel_launch(void* const* d_in, const int* in_sizes, int n_in,
                              void* d_out, int out_size, void* d_ws,
                              size_t ws_size, hipStream_t stream) {
  (void)in_sizes; (void)n_in; (void)out_size; (void)ws_size;
  const float* x      = (const float*)d_in[0];
  const float* keys   = (const float*)d_in[1];
  const float* values = (const float*)d_in[2];
  const float* wq     = (const float*)d_in[3];
  const float* wo     = (const float*)d_in[4];
  float* out = (float*)d_out;

  // Workspace (f16): Q [B,H,S,64] | ctx [B*S,512] | kh [B,H,S,64] | vh [B,H,64,S]
  const size_t N = (size_t)BATCH * HEADS * SEQ * KD;  // 4M elements each
  _Float16* q_ws   = (_Float16*)d_ws;
  _Float16* ctx_ws = q_ws + N;
  _Float16* kh_ws  = ctx_ws + N;
  _Float16* vh_ws  = kh_ws + N;

  kvprep_kernel<<<dim3(BATCH * HEADS * (SEQ / 64)), dim3(256), 0, stream>>>(
      keys, values, kh_ws, vh_ws);
  qproj_kernel<<<dim3(EMB / 16, (BATCH * SEQ) / 64), dim3(128), 0, stream>>>(
      x, wq, q_ws);
  attn_kernel<<<dim3(BATCH * HEADS * (SEQ / 64)), dim3(128), 0, stream>>>(
      kh_ws, vh_ws, q_ws, ctx_ws);
  oproj_kernel<<<dim3(EMB / 16, (BATCH * SEQ) / 64), dim3(128), 0, stream>>>(
      ctx_ws, wo, out);
}
